// TransformerBlock_687194767484
// MI455X (gfx1250) — matrix operand order repaired and checked by reading, not compiled, not run on hardware
//
#include <hip/hip_runtime.h>
#include <hip/hip_bf16.h>
#include <math.h>

// ---------------------------------------------------------------------------
// Point Transformer block for MI455X (gfx1250, wave32, WMMA).
// HBM-bound problem (~GBs of intermediates @ 23.3 TB/s >> math time), so:
//  * all big GEMMs use v_wmma_f32_16x16x32_f16 (f16 in, f32 accumulate)
//  * weights pre-packed once into B-operand lane layout (contiguous b128)
//  * large activations (x, h1, g, h2) are produced/stored directly in the
//    f16 A-operand fragment layout: halves HBM traffic, keeps A L2-resident,
//    and removes all f32->f16 converts from the GEMM inner loop.
// ---------------------------------------------------------------------------

typedef __attribute__((ext_vector_type(16))) _Float16 v16h;
typedef __attribute__((ext_vector_type(8)))  _Float16 v8h;
typedef __attribute__((ext_vector_type(8)))  float    v8f;

#define NBATCH 4
#define NPTS   1024   // points per batch (N-1)
#define DPTS   64
#define DMODEL 512
#define KNN    16
#define NROWS  (NBATCH*NPTS)        // 4096
#define NKROWS (NROWS*KNN)          // 65536
#define WSLOT  262144               // halves per packed-weight slot (512*512)

#define EPI_RELU  1
#define EPI_RESID 2

// A-fragment addressing (16x32 f16 tile striped over 32 lanes):
//   lane l (l<16):  rows mt*16+l,   K = kt*32 + {0..7, 16..23}
//   lane l (l>=16): rows mt*16+l-16,K = kt*32 + {8..15, 24..31}
//   16 halves contiguous per lane at  ((mt*(Kd/32)+kt32)*32 + l)*16

// ---------------------------------------------------------------------------
// Pack fp32 weight W[K x N] (row-major) into f16 B-operand fragments.
// ---------------------------------------------------------------------------
__global__ __launch_bounds__(256)
void pack_w_kernel(const float* __restrict__ W, _Float16* __restrict__ Wp,
                   int Kd, int N)
{
    const int lane = threadIdx.x & 31;
    const int wave = threadIdx.x >> 5;
    const int nt16 = N >> 4;
    const int kt32 = Kd >> 5;
    long tile = (long)blockIdx.x * 8 + wave;
    if (tile >= (long)kt32 * nt16) return;
    int kt = (int)(tile / nt16);
    int nt = (int)(tile % nt16);
    int col = nt * 16 + (lane & 15);
    int k0  = kt * 32 + (lane >> 4) * 8;
    _Float16* dst = Wp + (tile * 32 + lane) * 16;
#pragma unroll
    for (int i = 0; i < 16; ++i)
        dst[i] = (_Float16)W[(long)(k0 + i + (i >> 3) * 8) * N + col];
}

// repack fp32 activation A[M x Kd] into f16 A-fragment layout (one wave/frag)
__global__ __launch_bounds__(256)
void repack_a_kernel(const float* __restrict__ A, _Float16* __restrict__ Ap,
                     int M, int Kd)
{
    const int lane = threadIdx.x & 31;
    const int wave = threadIdx.x >> 5;
    const int kfr  = Kd >> 5;
    long frag = (long)blockIdx.x * 8 + wave;
    if (frag >= (long)(M >> 4) * kfr) return;
    int mt   = (int)(frag / kfr);
    int kt32 = (int)(frag % kfr);
    int row  = mt * 16 + (lane & 15);
    int k0   = kt32 * 32 + (lane >> 4) * 8;
    const float* src = A + (long)row * Kd + k0;
    v16h f;
#pragma unroll
    for (int i = 0; i < 8; ++i) f[i]     = (_Float16)src[i];
#pragma unroll
    for (int i = 0; i < 8; ++i) f[8 + i] = (_Float16)src[16 + i];
    *(v16h*)(Ap + (frag * 32 + lane) * 16) = f;
}

// ---------------------------------------------------------------------------
// GEMM: C = epilogue(A @ Wp + bias).  One wave -> 16x64 strip, 4 wmma / kstep.
// AF16: A already in packed f16 fragment layout.
// PACKOUT: write output as packed f16 A-fragments (LDS transpose) for next GEMM.
// ---------------------------------------------------------------------------
template <bool AF16, bool PACKOUT>
__global__ __launch_bounds__(256)
void gemm_wmma_kernel(const float* __restrict__ A32, const _Float16* __restrict__ A16,
                      const _Float16* __restrict__ Wp,
                      const float* __restrict__ bias, const float* __restrict__ resid,
                      float* __restrict__ C, _Float16* __restrict__ Cp,
                      int M, int Kd, int N, int flags)
{
    const int lane = threadIdx.x & 31;
    const int wave = threadIdx.x >> 5;
    const int tilesN = N >> 6;                 // 64-wide groups
    long tileId = (long)blockIdx.x * 8 + wave;
    int mt = (int)(tileId / tilesN);
    int ng = (int)(tileId % tilesN);
    if (mt * 16 >= M) return;                  // wave-uniform (grids sized exactly)

    const int half = lane >> 4;                // 0 | 1
    const int cl   = lane & 15;
    const int n0   = ng * 64;
    const int nt16 = N >> 4;
    const int kfr  = Kd >> 5;

    v8f acc[4] = {v8f{}, v8f{}, v8f{}, v8f{}};

    for (int kt = 0; kt < Kd; kt += 32) {
        v16h afrag;
        if (AF16) {
            const _Float16* ap = A16 + (((long)mt * kfr + (kt >> 5)) * 32 + lane) * 16;
            __builtin_prefetch(ap + 512, 0, 1);       // next fragment -> global_prefetch_b8
            afrag = *(const v16h*)ap;
        } else {
            const float* arow = A32 + (long)(mt * 16 + cl) * Kd + kt + half * 8;
            __builtin_prefetch(arow + 32, 0, 1);
#pragma unroll
            for (int i = 0; i < 8; ++i) afrag[i]     = (_Float16)arow[i];
#pragma unroll
            for (int i = 0; i < 8; ++i) afrag[8 + i] = (_Float16)arow[16 + i];
        }

        const _Float16* bbase =
            Wp + (((long)(kt >> 5) * nt16) + (n0 >> 4)) * 512 + lane * 16;
        v16h b0 = *(const v16h*)(bbase);
        v16h b1 = *(const v16h*)(bbase + 512);
        v16h b2 = *(const v16h*)(bbase + 1024);
        v16h b3 = *(const v16h*)(bbase + 1536);

        acc[0] = __builtin_amdgcn_wmma_f32_16x16x32_f16(false, afrag, false, b0,
                                                        (short)0, acc[0], false, false);
        acc[1] = __builtin_amdgcn_wmma_f32_16x16x32_f16(false, afrag, false, b1,
                                                        (short)0, acc[1], false, false);
        acc[2] = __builtin_amdgcn_wmma_f32_16x16x32_f16(false, afrag, false, b2,
                                                        (short)0, acc[2], false, false);
        acc[3] = __builtin_amdgcn_wmma_f32_16x16x32_f16(false, afrag, false, b3,
                                                        (short)0, acc[3], false, false);
    }

    if (PACKOUT) {
        // Transpose the wave's 16x64 tile into next-GEMM A-fragment layout via
        // per-wave LDS staging (rows padded to 80 halves = 160B, 16B aligned).
        __shared__ _Float16 sh[8][16][80];
        _Float16 (*s)[80] = sh[wave];
#pragma unroll
        for (int j = 0; j < 4; ++j) {
#pragma unroll
            for (int i = 0; i < 8; ++i) {
                float v = acc[j][i];
                if (bias) v += bias[n0 + j * 16 + cl];
                if (flags & EPI_RELU) v = fmaxf(v, 0.0f);
                s[half * 8 + i][j * 16 + cl] = (_Float16)v;
            }
        }
        // same-wave RAW through LDS; compiler inserts s_wait_dscnt
#pragma unroll
        for (int kc = 0; kc < 2; ++kc) {               // 64 cols = 2 K-chunks of 32
            int kt32 = (n0 >> 5) + kc;
            int c0 = kc * 32 + half * 8;
            v8h lo = *(const v8h*)&s[cl][c0];
            v8h hi = *(const v8h*)&s[cl][c0 + 16];
            v16h frag = __builtin_shufflevector(lo, hi, 0, 1, 2, 3, 4, 5, 6, 7,
                                                8, 9, 10, 11, 12, 13, 14, 15);
            *(v16h*)(Cp + (((long)mt * (N >> 5) + kt32) * 32 + lane) * 16) = frag;
        }
    } else {
#pragma unroll
        for (int j = 0; j < 4; ++j) {
#pragma unroll
            for (int i = 0; i < 8; ++i) {
                int r = mt * 16 + half * 8 + i;
                int c = n0 + j * 16 + cl;
                float v = acc[j][i];
                if (bias) v += bias[c];
                if (flags & EPI_RELU)  v = fmaxf(v, 0.0f);
                if (flags & EPI_RESID) v += resid[(long)r * N + c];
                C[(long)r * N + c] = v;
            }
        }
    }
}

// ---------------------------------------------------------------------------
// KNN: one block per (b,m); brute-force distances + 16x argmin selection
// (stable tie-break on smallest index, matching argsort).
// ---------------------------------------------------------------------------
__global__ __launch_bounds__(256)
void knn_kernel(const float* __restrict__ xyz, int* __restrict__ knn)
{
    __shared__ float sd[NPTS];
    __shared__ float rv[256];
    __shared__ int   ri[256];
    const int bm = blockIdx.x;
    const int b = bm >> 10, m = bm & 1023;
    const int tid = threadIdx.x;
    const float* base = xyz + ((long)b * (NPTS + 1) + 1) * 3;
    const float qx = base[m * 3 + 0], qy = base[m * 3 + 1], qz = base[m * 3 + 2];
    for (int n = tid; n < NPTS; n += 256) {
        float dx = base[n * 3 + 0] - qx;
        float dy = base[n * 3 + 1] - qy;
        float dz = base[n * 3 + 2] - qz;
        sd[n] = dx * dx + dy * dy + dz * dz;
    }
    __syncthreads();
    for (int kk = 0; kk < KNN; ++kk) {
        float best = 1e30f; int bi = NPTS;
        for (int n = tid; n < NPTS; n += 256) {
            float d = sd[n];
            if (d < best || (d == best && n < bi)) { best = d; bi = n; }
        }
        rv[tid] = best; ri[tid] = bi;
        __syncthreads();
        for (int s = 128; s > 0; s >>= 1) {
            if (tid < s) {
                if (rv[tid + s] < rv[tid] ||
                    (rv[tid + s] == rv[tid] && ri[tid + s] < ri[tid])) {
                    rv[tid] = rv[tid + s]; ri[tid] = ri[tid + s];
                }
            }
            __syncthreads();
        }
        if (tid == 0) { knn[bm * KNN + kk] = ri[0]; sd[ri[0]] = 1e30f; }
        __syncthreads();
    }
}

// gather features[:,1:,:] -> contiguous 4096x64
__global__ void gather_feats_kernel(const float* __restrict__ features,
                                    float* __restrict__ feats_c)
{
    long i = (long)blockIdx.x * blockDim.x + threadIdx.x;
    if (i >= (long)NROWS * DPTS) return;
    int row = (int)(i >> 6), j = (int)(i & 63);
    int b = row >> 10, m = row & 1023;
    feats_c[i] = features[((long)b * (NPTS + 1) + 1 + m) * DPTS + j];
}

// h1 = relu((p_m - p_knn) @ delta1 + b1), emitted directly as packed f16 A.
// One wave per fragment (mt, kt32); Kd = DMODEL.
__global__ __launch_bounds__(256)
void build_h1_pack_kernel(const float* __restrict__ xyz, const int* __restrict__ knn,
                          const float* __restrict__ d1w, const float* __restrict__ d1b,
                          _Float16* __restrict__ h1p)
{
    const int lane = threadIdx.x & 31;
    const int wave = threadIdx.x >> 5;
    const int kfr = DMODEL >> 5;               // 16
    long frag = (long)blockIdx.x * 8 + wave;   // < (NKROWS/16)*16 = 65536
    int mt   = (int)(frag / kfr);
    int kt32 = (int)(frag % kfr);
    const int rowg = mt * 16 + (lane & 15);    // (b*1024+m)*16+kn
    const int b = rowg >> 14, rem = rowg & 16383;
    const int m = rem >> 4;
    const int idx = knn[rowg];
    const float* base = xyz + ((long)b * (NPTS + 1) + 1) * 3;
    const float d0 = base[m * 3 + 0] - base[idx * 3 + 0];
    const float d1 = base[m * 3 + 1] - base[idx * 3 + 1];
    const float d2 = base[m * 3 + 2] - base[idx * 3 + 2];
    const int k0 = kt32 * 32 + (lane >> 4) * 8;
    v16h f;
#pragma unroll
    for (int u = 0; u < 2; ++u) {              // two contiguous 8-runs: k0, k0+16
        int kb = k0 + u * 16;
#pragma unroll
        for (int i = 0; i < 8; ++i) {
            int k = kb + i;
            float v = d0 * d1w[k] + d1 * d1w[DMODEL + k] + d2 * d1w[2 * DMODEL + k] + d1b[k];
            f[u * 8 + i] = (_Float16)fmaxf(v, 0.0f);
        }
    }
    *(v16h*)(h1p + (frag * 32 + lane) * 16) = f;
}

// g = q[m] - k[knn] + pos, emitted directly as packed f16 A.
__global__ __launch_bounds__(256)
void build_g_pack_kernel(const float* __restrict__ q, const float* __restrict__ kp,
                         const float* __restrict__ pos, const int* __restrict__ knn,
                         _Float16* __restrict__ gp)
{
    const int lane = threadIdx.x & 31;
    const int wave = threadIdx.x >> 5;
    const int kfr = DMODEL >> 5;
    long frag = (long)blockIdx.x * 8 + wave;
    int mt   = (int)(frag / kfr);
    int kt32 = (int)(frag % kfr);
    const int rowg = mt * 16 + (lane & 15);
    const int b = rowg >> 14, rem = rowg & 16383;
    const int m = rem >> 4;
    const int idx = knn[rowg];
    const float* qr = q   + ((long)b * NPTS + m)   * DMODEL;
    const float* kr = kp  + ((long)b * NPTS + idx) * DMODEL;
    const float* pr = pos + (long)rowg * DMODEL;
    const int k0 = kt32 * 32 + (lane >> 4) * 8;
    v16h f;
#pragma unroll
    for (int u = 0; u < 2; ++u) {
        int kb = k0 + u * 16;
#pragma unroll
        for (int i = 0; i < 8; ++i) {
            int k = kb + i;
            f[u * 8 + i] = (_Float16)(qr[k] - kr[k] + pr[k]);
        }
    }
    *(v16h*)(gp + (frag * 32 + lane) * 16) = f;
}

// softmax over the K=16 neighbor axis, per (b,m,f); IN PLACE (regs first).
__global__ __launch_bounds__(256)
void softmax_k_kernel(float* __restrict__ attn)
{
    const int bm = blockIdx.x;
    const long base0 = (long)bm * KNN * DMODEL;
    const float scale = 0.044194173824159216f;   // 1/sqrt(512)
    for (int f = threadIdx.x; f < DMODEL; f += 256) {
        float vals[KNN];
        float mx = -1e30f;
#pragma unroll
        for (int k = 0; k < KNN; ++k) {
            vals[k] = attn[base0 + (long)k * DMODEL + f] * scale;
            mx = fmaxf(mx, vals[k]);
        }
        float sum = 0.0f;
#pragma unroll
        for (int k = 0; k < KNN; ++k) { vals[k] = __expf(vals[k] - mx); sum += vals[k]; }
        float inv = 1.0f / sum;
#pragma unroll
        for (int k = 0; k < KNN; ++k)
            attn[base0 + (long)k * DMODEL + f] = vals[k] * inv;
    }
}

// res[b,m,f] = sum_k attn * (v[knn] + pos)
__global__ __launch_bounds__(256)
void reduce_res_kernel(const float* __restrict__ attn, const float* __restrict__ vp,
                       const float* __restrict__ pos, const int* __restrict__ knn,
                       float* __restrict__ res)
{
    __shared__ int sidx[KNN];
    const int bm = blockIdx.x;
    const int b = bm >> 10;
    if (threadIdx.x < KNN) sidx[threadIdx.x] = knn[bm * KNN + threadIdx.x];
    __syncthreads();
    const long base0 = (long)bm * KNN * DMODEL;
    for (int f = threadIdx.x; f < DMODEL; f += 256) {
        float acc = 0.0f;
#pragma unroll
        for (int k = 0; k < KNN; ++k) {
            float a = attn[base0 + (long)k * DMODEL + f];
            float v = vp[((long)b * NPTS + sidx[k]) * DMODEL + f]
                    + pos[base0 + (long)k * DMODEL + f];
            acc += a * v;
        }
        res[(long)bm * DMODEL + f] = acc;
    }
}

// scatter 4096x64 point results into out[b,1+m,:]
__global__ void scatter_out_kernel(const float* __restrict__ res2, float* __restrict__ out)
{
    long i = (long)blockIdx.x * blockDim.x + threadIdx.x;
    if (i >= (long)NROWS * DPTS) return;
    int row = (int)(i >> 6), j = (int)(i & 63);
    int b = row >> 10, m = row & 1023;
    out[((long)b * (NPTS + 1) + 1 + m) * DPTS + j] = res2[i];
}

// ------------------------------- CLS path (B=4, tiny) ----------------------
__global__ __launch_bounds__(512)
void cls_proj_kernel(const float* __restrict__ features, const float* __restrict__ w1,
                     const float* __restrict__ b1, const float* __restrict__ wqc,
                     float* __restrict__ cx, float* __restrict__ qc)
{
    __shared__ float scls[DPTS];
    __shared__ float scx[DMODEL];
    const int b = blockIdx.x, f = threadIdx.x;
    if (f < DPTS) scls[f] = features[(long)b * (NPTS + 1) * DPTS + f];
    __syncthreads();
    float acc = b1[f];
    for (int i = 0; i < DPTS; ++i) acc += scls[i] * w1[i * DMODEL + f];
    scx[f] = acc; cx[b * DMODEL + f] = acc;
    __syncthreads();
    float aq = 0.0f;
    for (int i = 0; i < DMODEL; ++i) aq += scx[i] * wqc[i * DMODEL + f];
    qc[b * DMODEL + f] = aq;
}

__global__ __launch_bounds__(256)
void cls_attn_kernel(const float* __restrict__ qc, const float* __restrict__ kc,
                     float* __restrict__ cls_attn)
{
    __shared__ float sq[DMODEL];
    __shared__ float sc[NPTS];
    __shared__ float red[256];
    const int b = blockIdx.x, tid = threadIdx.x;
    for (int f = tid; f < DMODEL; f += 256) sq[f] = qc[b * DMODEL + f];
    __syncthreads();
    const float scale = 0.044194173824159216f;
    for (int n = tid; n < NPTS; n += 256) {
        const float* kr = kc + ((long)b * NPTS + n) * DMODEL;
        float acc = 0.0f;
        for (int f = 0; f < DMODEL; ++f) acc += sq[f] * kr[f];
        sc[n] = acc * scale;
    }
    __syncthreads();
    float mx = -1e30f;
    for (int n = tid; n < NPTS; n += 256) mx = fmaxf(mx, sc[n]);
    red[tid] = mx; __syncthreads();
    for (int s = 128; s; s >>= 1) { if (tid < s) red[tid] = fmaxf(red[tid], red[tid + s]); __syncthreads(); }
    mx = red[0]; __syncthreads();
    float sum = 0.0f;
    for (int n = tid; n < NPTS; n += 256) { float e = __expf(sc[n] - mx); sc[n] = e; sum += e; }
    red[tid] = sum; __syncthreads();
    for (int s = 128; s; s >>= 1) { if (tid < s) red[tid] += red[tid + s]; __syncthreads(); }
    const float inv = 1.0f / red[0];
    for (int n = tid; n < NPTS; n += 256) cls_attn[b * NPTS + n] = sc[n] * inv;
}

__global__ __launch_bounds__(512)
void cls_res_kernel(const float* __restrict__ cls_attn, const float* __restrict__ vc,
                    const float* __restrict__ w2, const float* __restrict__ b2,
                    const float* __restrict__ features, float* __restrict__ out)
{
    __shared__ float st[DMODEL];
    const int b = blockIdx.x, f = threadIdx.x;
    float acc = 0.0f;
    for (int n = 0; n < NPTS; ++n)
        acc += cls_attn[b * NPTS + n] * vc[((long)b * NPTS + n) * DMODEL + f];
    st[f] = acc;
    __syncthreads();
    if (f < DPTS) {
        float o = b2[f];
        for (int i = 0; i < DMODEL; ++i) o += st[i] * w2[i * DPTS + f];
        o += features[(long)b * (NPTS + 1) * DPTS + f];
        out[(long)b * (NPTS + 1) * DPTS + f] = o;
    }
}

// ---------------------------------------------------------------------------
extern "C" void kernel_launch(void* const* d_in, const int* in_sizes, int n_in,
                              void* d_out, int out_size, void* d_ws, size_t ws_size,
                              hipStream_t stream)
{
    (void)in_sizes; (void)n_in; (void)out_size; (void)ws_size;
    const float* xyz      = (const float*)d_in[0];
    const float* features = (const float*)d_in[1];
    const float* fc1_w    = (const float*)d_in[2];
    const float* fc1_b    = (const float*)d_in[3];
    const float* fc2_w    = (const float*)d_in[4];
    const float* fc2_b    = (const float*)d_in[5];
    const float* d1w      = (const float*)d_in[6];
    const float* d1b      = (const float*)d_in[7];
    const float* d2w      = (const float*)d_in[8];
    const float* d2b      = (const float*)d_in[9];
    const float* g1w      = (const float*)d_in[10];
    const float* g1b      = (const float*)d_in[11];
    const float* g2w      = (const float*)d_in[12];
    const float* g2b      = (const float*)d_in[13];
    const float* wq       = (const float*)d_in[14];
    const float* wk       = (const float*)d_in[15];
    const float* wv       = (const float*)d_in[16];
    const float* fc1c_w   = (const float*)d_in[17];
    const float* fc1c_b   = (const float*)d_in[18];
    const float* fc2c_w   = (const float*)d_in[19];
    const float* fc2c_b   = (const float*)d_in[20];
    const float* wqc      = (const float*)d_in[21];
    const float* wkc      = (const float*)d_in[22];
    const float* wvc      = (const float*)d_in[23];

    // d_out layout: out(4*1025*64) | attn(4*1024*16*512) | cls_attn(4*1024)
    float* out_pt  = (float*)d_out;
    float* attn_pt = out_pt + (long)NBATCH * (NPTS + 1) * DPTS;
    float* clsa_pt = attn_pt + (long)NKROWS * DMODEL;

    // workspace carve-out
    char* ws = (char*)d_ws;
    size_t off = 0;
    auto take = [&](size_t bytes) { size_t o = off; off += (bytes + 255) & ~(size_t)255; return o; };
    _Float16* wp   = (_Float16*)(ws + take(10 * (size_t)WSLOT * sizeof(_Float16)));
    int*   knn     = (int*)     (ws + take((size_t)NKROWS * sizeof(int)));
    float* feats_c = (float*)   (ws + take((size_t)NROWS * DPTS * 4));
    float* x       = (float*)   (ws + take((size_t)NROWS * DMODEL * 4));
    _Float16* xp   = (_Float16*)(ws + take((size_t)NROWS * DMODEL * 2));
    float* q       = (float*)   (ws + take((size_t)NROWS * DMODEL * 4));
    float* kp      = (float*)   (ws + take((size_t)NROWS * DMODEL * 4));
    float* vp      = (float*)   (ws + take((size_t)NROWS * DMODEL * 4));
    float* kc      = (float*)   (ws + take((size_t)NROWS * DMODEL * 4));
    float* vc      = (float*)   (ws + take((size_t)NROWS * DMODEL * 4));
    _Float16* h1gp = (_Float16*)(ws + take((size_t)NKROWS * DMODEL * 2)); // h1 then g
    _Float16* h2p  = (_Float16*)(ws + take((size_t)NKROWS * DMODEL * 2)); // gamma1 out
    float* posb    = (float*)   (ws + take((size_t)NKROWS * DMODEL * 4));
    float* res     = (float*)   (ws + take((size_t)NROWS * DMODEL * 4));
    float* res2    = (float*)   (ws + take((size_t)NROWS * DPTS * 4));
    float* cx      = (float*)   (ws + take((size_t)NBATCH * DMODEL * 4));
    float* qcv     = (float*)   (ws + take((size_t)NBATCH * DMODEL * 4));

    _Float16* p_fc1 = wp + 0 * WSLOT;
    _Float16* p_wq  = wp + 1 * WSLOT;
    _Float16* p_wk  = wp + 2 * WSLOT;
    _Float16* p_wv  = wp + 3 * WSLOT;
    _Float16* p_wkc = wp + 4 * WSLOT;
    _Float16* p_wvc = wp + 5 * WSLOT;
    _Float16* p_d2  = wp + 6 * WSLOT;
    _Float16* p_g1  = wp + 7 * WSLOT;
    _Float16* p_g2  = wp + 8 * WSLOT;
    _Float16* p_fc2 = wp + 9 * WSLOT;

    auto pack = [&](const float* W, _Float16* Wpk, int Kd, int N) {
        int tiles = (Kd >> 5) * (N >> 4);
        pack_w_kernel<<<(tiles + 7) / 8, 256, 0, stream>>>(W, Wpk, Kd, N);
    };
    auto gemm32 = [&](const float* A, const _Float16* Wpk, const float* bias,
                      const float* resid, float* C, int M, int Kd, int N, int flags) {
        long tiles = (long)(M / 16) * (N / 64);
        gemm_wmma_kernel<false, false><<<(int)((tiles + 7) / 8), 256, 0, stream>>>(
            A, nullptr, Wpk, bias, resid, C, nullptr, M, Kd, N, flags);
    };
    auto gemm16 = [&](const _Float16* Ap, const _Float16* Wpk, const float* bias,
                      float* C, int M, int Kd, int N, int flags) {
        long tiles = (long)(M / 16) * (N / 64);
        gemm_wmma_kernel<true, false><<<(int)((tiles + 7) / 8), 256, 0, stream>>>(
            nullptr, Ap, Wpk, bias, nullptr, C, nullptr, M, Kd, N, flags);
    };
    auto gemm16p = [&](const _Float16* Ap, const _Float16* Wpk, const float* bias,
                       _Float16* Cpk, int M, int Kd, int N, int flags) {
        long tiles = (long)(M / 16) * (N / 64);
        gemm_wmma_kernel<true, true><<<(int)((tiles + 7) / 8), 256, 0, stream>>>(
            nullptr, Ap, Wpk, bias, nullptr, nullptr, Cpk, M, Kd, N, flags);
    };

    // 1) pack all WMMA weights (f32 -> f16 fragment layout)
    pack(fc1_w, p_fc1, DPTS,  DMODEL);
    pack(wq,    p_wq,  DMODEL, DMODEL);
    pack(wk,    p_wk,  DMODEL, DMODEL);
    pack(wv,    p_wv,  DMODEL, DMODEL);
    pack(wkc,   p_wkc, DMODEL, DMODEL);
    pack(wvc,   p_wvc, DMODEL, DMODEL);
    pack(d2w,   p_d2,  DMODEL, DMODEL);
    pack(g1w,   p_g1,  DMODEL, DMODEL);
    pack(g2w,   p_g2,  DMODEL, DMODEL);
    pack(fc2_w, p_fc2, DMODEL, DPTS);

    // 2) gather point features; 3) brute-force KNN
    gather_feats_kernel<<<(NROWS * DPTS + 255) / 256, 256, 0, stream>>>(features, feats_c);
    knn_kernel<<<NROWS, 256, 0, stream>>>(xyz, knn);

    // 4) x = feats @ fc1 + b, then repack x to f16 A-fragments
    gemm32(feats_c, p_fc1, fc1_b, nullptr, x, NROWS, DPTS, DMODEL, 0);
    {
        int frags = (NROWS / 16) * (DMODEL / 32);
        repack_a_kernel<<<(frags + 7) / 8, 256, 0, stream>>>(x, xp, NROWS, DMODEL);
    }

    // 5) q/k/v/kc/vc projections (f16-A fast path)
    gemm16(xp, p_wq,  nullptr, q,  NROWS, DMODEL, DMODEL, 0);
    gemm16(xp, p_wk,  nullptr, kp, NROWS, DMODEL, DMODEL, 0);
    gemm16(xp, p_wv,  nullptr, vp, NROWS, DMODEL, DMODEL, 0);
    gemm16(xp, p_wkc, nullptr, kc, NROWS, DMODEL, DMODEL, 0);
    gemm16(xp, p_wvc, nullptr, vc, NROWS, DMODEL, DMODEL, 0);

    // 6) pos-enc MLP: packed f16 h1, then 65536x512x512 WMMA GEMM -> pos (f32)
    {
        int frags = (NKROWS / 16) * (DMODEL / 32);
        build_h1_pack_kernel<<<(frags + 7) / 8, 256, 0, stream>>>(xyz, knn, d1w, d1b, h1gp);
    }
    gemm16(h1gp, p_d2, d2b, posb, NKROWS, DMODEL, DMODEL, 0);

    // 7) gamma MLP: g (packed f16, reuses h1 buffer) -> relu GEMM with packed
    //    f16 output (LDS transpose epilogue) -> GEMM into d_out attn region
    {
        int frags = (NKROWS / 16) * (DMODEL / 32);
        build_g_pack_kernel<<<(frags + 7) / 8, 256, 0, stream>>>(q, kp, posb, knn, h1gp);
    }
    gemm16p(h1gp, p_g1, g1b, h2p, NKROWS, DMODEL, DMODEL, EPI_RELU);
    gemm16(h2p, p_g2, g2b, attn_pt, NKROWS, DMODEL, DMODEL, 0);

    // 8) softmax over K (in place in d_out); 9) weighted reduction
    softmax_k_kernel<<<NROWS, 256, 0, stream>>>(attn_pt);
    reduce_res_kernel<<<NROWS, 256, 0, stream>>>(attn_pt, vp, posb, knn, res);

    // 10) res @ fc2 + b + feats residual, scatter into out[:,1:,:]
    gemm32(res, p_fc2, fc2_b, feats_c, res2, NROWS, DMODEL, DPTS, EPI_RESID);
    scatter_out_kernel<<<(NROWS * DPTS + 255) / 256, 256, 0, stream>>>(res2, out_pt);

    // 11) CLS token path (B=4, latency-trivial)
    cls_proj_kernel<<<NBATCH, DMODEL, 0, stream>>>(features, fc1c_w, fc1c_b, wqc, cx, qcv);
    cls_attn_kernel<<<NBATCH, 256, 0, stream>>>(qcv, kc, clsa_pt);
    cls_res_kernel<<<NBATCH, DMODEL, 0, stream>>>(clsa_pt, vc, fc2c_w, fc2c_b, features, out_pt);
}
